// CfCAttention_77352361001016
// MI455X (gfx1250) — compile-verified, matrix operand order checked by reading, not run on hardware
//
#include <hip/hip_runtime.h>
#include <hip/hip_bf16.h>

// ---------------------------------------------------------------------------
// CfC layer for MI455X (gfx1250, wave32, WMMA).
//   B=4, T=4096, C=1024.  Three GEMMs (16384x1024 @ 1024x1024) + linear scan.
//   GEMMs: bf16 WMMA (v_wmma_f32_16x16x32_bf16), fp32 accumulate.
//   Staging: triple-buffered GLOBAL_LOAD_ASYNC_TO_LDS_B128 with a nonzero
//   s_wait_asynccnt threshold -> one stage always in flight behind compute.
// ---------------------------------------------------------------------------

typedef __attribute__((ext_vector_type(16))) __bf16 v16bf;
typedef __attribute__((ext_vector_type(8)))  float  v8f;

union Frag16 { v16bf v; uint4 q[2]; };

#define CDIM 1024
#define MROWS 16384   // B*T
#define BM 64
#define BN 128
#define BK 32
#define PADK 40       // padded LDS row stride (80B) -> conflict-free b128 reads

// async 16B global->LDS copy; ldsoff = byte offset in workgroup LDS segment
#define ASYNC_B128(ldsoff, gptr)                                         \
  asm volatile("global_load_async_to_lds_b128 %0, %1, off"               \
               :: "v"(ldsoff), "v"(gptr) : "memory")
// wait until at most `n` async ops remain outstanding on this wave
#define WAIT_ASYNC_LE(n)                                                 \
  asm volatile("s_wait_asynccnt %0" :: "i"(n) : "memory")

// ---------------------------------------------------------------------------
// fp32 -> bf16 conversion, 4 elements/thread, 8B stores.
// ---------------------------------------------------------------------------
__global__ __launch_bounds__(256) void cvt_f32_bf16(const float* __restrict__ in,
                                                    __bf16* __restrict__ out, int n) {
  int i = (blockIdx.x * 256 + threadIdx.x) * 4;
  if (i + 3 < n) {
    float4 v = *(const float4*)(in + i);
    union { __bf16 h[4]; uint2 u; } p;
    p.h[0] = (__bf16)v.x; p.h[1] = (__bf16)v.y;
    p.h[2] = (__bf16)v.z; p.h[3] = (__bf16)v.w;
    *(uint2*)(out + i) = p.u;
  }
}

// ---------------------------------------------------------------------------
// Fused gate GEMM: f = sigmoid(x Wf^T), g = tanh(x Wg^T)
//   Shared x tile feeds both weight tiles; epilogue emits a=f, b=(1-f)g (bf16).
// 256 threads = 8 waves, 2(M) x 4(N); wave tile 32x32 = 2x2 fragments/gate.
// Dynamic LDS per buffer: [A 5120B][F 10240B][G 10240B], 3 buffers (76.8KB).
// ---------------------------------------------------------------------------
#define G_BUF_BYTES 25600u
#define G_A_OFF 0u
#define G_F_OFF 5120u
#define G_G_OFF 15360u
#define G_STAGE_ASYNCS 5

__global__ __launch_bounds__(256)
void gemm_gates(const __bf16* __restrict__ X, const __bf16* __restrict__ Wf,
                const __bf16* __restrict__ Wg, __bf16* __restrict__ Abuf,
                __bf16* __restrict__ Bbuf) {
  extern __shared__ char smem[];
  const __bf16* sbase = (const __bf16*)smem;

  const int tid  = threadIdx.x;
  const int lane = tid & 31;
  const int w    = tid >> 5;
  const int wm   = w & 1;        // 0..1  (M waves)
  const int wn   = w >> 1;       // 0..3  (N waves)
  const int lrow = lane & 15;
  const int lhi  = lane >> 4;    // K-half selector (ISA 16-bit A/B layout)
  const long m0  = (long)blockIdx.x * BM;
  const long n0  = (long)blockIdx.y * BN;

  // per-thread staging coordinates (16B chunks)
  const int arow = tid >> 2, acp = tid & 3;            // A: 64 rows x 4 chunks
  const unsigned aoff = (unsigned)(arow * PADK + acp * 8) * 2u;

  const v8f vz = {0.f, 0.f, 0.f, 0.f, 0.f, 0.f, 0.f, 0.f};
  v8f cf[2][2], cg[2][2];
#pragma unroll
  for (int i = 0; i < 2; ++i)
#pragma unroll
    for (int j = 0; j < 2; ++j) { cf[i][j] = vz; cg[i][j] = vz; }

  // stage K-tile k0 into LDS buffer `buf` (5 async b128 per thread/wave)
  auto stage = [&](int buf, int k0) {
    unsigned base = (unsigned)buf * G_BUF_BYTES;
    ASYNC_B128(base + G_A_OFF + aoff, &X[(m0 + arow) * CDIM + k0 + acp * 8]);
#pragma unroll
    for (int i = 0; i < 2; ++i) {                      // B: 128 rows x 4 chunks
      int chunk = tid + i * 256;
      int r = chunk >> 2, cp = chunk & 3;
      unsigned boff = (unsigned)(r * PADK + cp * 8) * 2u;
      ASYNC_B128(base + G_F_OFF + boff, &Wf[(n0 + r) * CDIM + k0 + cp * 8]);
      ASYNC_B128(base + G_G_OFF + boff, &Wg[(n0 + r) * CDIM + k0 + cp * 8]);
    }
  };

  // prologue: two stages in flight, wait for the oldest only
  stage(0, 0);
  stage(1, BK);
  WAIT_ASYNC_LE(G_STAGE_ASYNCS);
  __syncthreads();

  int buf = 0;
  for (int k0 = 0; k0 < CDIM; k0 += BK) {
    // issue stage for tile k0+2*BK (wrapped; over-fetch never consumed)
    int bnext = buf + 2; if (bnext >= 3) bnext -= 3;
    stage(bnext, (k0 + 2 * BK) & (CDIM - 1));

    const __bf16* sA = sbase + (unsigned)buf * (G_BUF_BYTES / 2) + G_A_OFF / 2;
    const __bf16* sF = sbase + (unsigned)buf * (G_BUF_BYTES / 2) + G_F_OFF / 2;
    const __bf16* sG = sbase + (unsigned)buf * (G_BUF_BYTES / 2) + G_G_OFF / 2;

    Frag16 fa[2], fbf[2], fbg[2];
#pragma unroll
    for (int i = 0; i < 2; ++i) {
      int r = wm * 32 + i * 16 + lrow;
      fa[i].q[0] = *(const uint4*)&sA[r * PADK + lhi * 8];
      fa[i].q[1] = *(const uint4*)&sA[r * PADK + 16 + lhi * 8];
    }
#pragma unroll
    for (int j = 0; j < 2; ++j) {
      int r = wn * 32 + j * 16 + lrow;
      fbf[j].q[0] = *(const uint4*)&sF[r * PADK + lhi * 8];
      fbf[j].q[1] = *(const uint4*)&sF[r * PADK + 16 + lhi * 8];
      fbg[j].q[0] = *(const uint4*)&sG[r * PADK + lhi * 8];
      fbg[j].q[1] = *(const uint4*)&sG[r * PADK + 16 + lhi * 8];
    }
#pragma unroll
    for (int i = 0; i < 2; ++i)
#pragma unroll
      for (int j = 0; j < 2; ++j) {
        cf[i][j] = __builtin_amdgcn_wmma_f32_16x16x32_bf16(
            false, fa[i].v, false, fbf[j].v, (short)0, cf[i][j], false, false);
        cg[i][j] = __builtin_amdgcn_wmma_f32_16x16x32_bf16(
            false, fa[i].v, false, fbg[j].v, (short)0, cg[i][j], false, false);
      }

    // stage k0+BK complete (only the newest stage may remain in flight)
    WAIT_ASYNC_LE(G_STAGE_ASYNCS);
    __syncthreads();
    buf = buf + 1; if (buf >= 3) buf -= 3;
  }

  // ---- epilogue: gates + scan coefficients ----
#pragma unroll
  for (int i = 0; i < 2; ++i)
#pragma unroll
    for (int j = 0; j < 2; ++j)
#pragma unroll
      for (int e = 0; e < 8; ++e) {
        long row = m0 + wm * 32 + i * 16 + lhi * 8 + e;
        long col = n0 + wn * 32 + j * 16 + lrow;
        float f = 1.f / (1.f + expf(-cf[i][j][e]));
        float g = tanhf(cg[i][j][e]);
        Abuf[row * CDIM + col] = (__bf16)f;
        Bbuf[row * CDIM + col] = (__bf16)((1.f - f) * g);
      }
}

// ---------------------------------------------------------------------------
// Chunked parallel scan of h_t = a_t*h_{t-1} + b_t over T=4096 (64 chunks x 64).
// Affine composition: over a chunk  h_out = P*h_in + Q.
// ---------------------------------------------------------------------------
#define NCOLS 4096   // B*C independent sequences
#define NCHUNK 64
#define CHLEN 64

__global__ __launch_bounds__(256)
void scan_pass1(const __bf16* __restrict__ a, const __bf16* __restrict__ b,
                float* __restrict__ aggP, float* __restrict__ aggQ) {
  int gid = blockIdx.x * 256 + threadIdx.x;    // 0 .. 256K-1
  int col = gid & (NCOLS - 1);
  int ch  = gid >> 12;
  int bI = col >> 10, c = col & (CDIM - 1);
  long base = ((long)bI * 4096 + ch * CHLEN) * CDIM + c;
  float P = 1.f, Q = 0.f;
  for (int t = 0; t < CHLEN; ++t) {
    float at = (float)a[base + (long)t * CDIM];
    float bt = (float)b[base + (long)t * CDIM];
    P = at * P;
    Q = at * Q + bt;
  }
  aggP[(long)ch * NCOLS + col] = P;
  aggQ[(long)ch * NCOLS + col] = Q;
}

__global__ __launch_bounds__(256)
void scan_pass2(const float* __restrict__ aggP, const float* __restrict__ aggQ,
                float* __restrict__ carry) {
  int col = blockIdx.x * 256 + threadIdx.x;    // 0 .. 4095
  float h = 0.f;
  for (int ch = 0; ch < NCHUNK; ++ch) {
    long idx = (long)ch * NCOLS + col;
    carry[idx] = h;                             // carry-in for this chunk
    h = aggP[idx] * h + aggQ[idx];
  }
}

__global__ __launch_bounds__(256)
void scan_pass3(const __bf16* __restrict__ a, const __bf16* __restrict__ b,
                const float* __restrict__ carry, __bf16* __restrict__ h) {
  int gid = blockIdx.x * 256 + threadIdx.x;
  int col = gid & (NCOLS - 1);
  int ch  = gid >> 12;
  int bI = col >> 10, c = col & (CDIM - 1);
  long base = ((long)bI * 4096 + ch * CHLEN) * CDIM + c;
  float hv = carry[(long)ch * NCOLS + col];
  for (int t = 0; t < CHLEN; ++t) {
    float at = (float)a[base + (long)t * CDIM];
    float bt = (float)b[base + (long)t * CDIM];
    hv = at * hv + bt;
    h[base + (long)t * CDIM] = (__bf16)hv;
  }
}

// ---------------------------------------------------------------------------
// Projection GEMM: out = h Wp^T  (fp32 output straight to d_out)
// Dynamic LDS per buffer: [A 5120B][B 10240B], 3 buffers (46KB).
// ---------------------------------------------------------------------------
#define P_BUF_BYTES 15360u
#define P_A_OFF 0u
#define P_B_OFF 5120u
#define P_STAGE_ASYNCS 3

__global__ __launch_bounds__(256)
void gemm_proj(const __bf16* __restrict__ H, const __bf16* __restrict__ Wp,
               float* __restrict__ Out) {
  extern __shared__ char smem[];
  const __bf16* sbase = (const __bf16*)smem;

  const int tid  = threadIdx.x;
  const int lane = tid & 31;
  const int w    = tid >> 5;
  const int wm   = w & 1;
  const int wn   = w >> 1;
  const int lrow = lane & 15;
  const int lhi  = lane >> 4;
  const long m0  = (long)blockIdx.x * BM;
  const long n0  = (long)blockIdx.y * BN;

  const int arow = tid >> 2, acp = tid & 3;
  const unsigned aoff = (unsigned)(arow * PADK + acp * 8) * 2u;

  const v8f vz = {0.f, 0.f, 0.f, 0.f, 0.f, 0.f, 0.f, 0.f};
  v8f acc[2][2];
#pragma unroll
  for (int i = 0; i < 2; ++i)
#pragma unroll
    for (int j = 0; j < 2; ++j) acc[i][j] = vz;

  auto stage = [&](int buf, int k0) {
    unsigned base = (unsigned)buf * P_BUF_BYTES;
    ASYNC_B128(base + P_A_OFF + aoff, &H[(m0 + arow) * CDIM + k0 + acp * 8]);
#pragma unroll
    for (int i = 0; i < 2; ++i) {
      int chunk = tid + i * 256;
      int r = chunk >> 2, cp = chunk & 3;
      unsigned boff = (unsigned)(r * PADK + cp * 8) * 2u;
      ASYNC_B128(base + P_B_OFF + boff, &Wp[(n0 + r) * CDIM + k0 + cp * 8]);
    }
  };

  stage(0, 0);
  stage(1, BK);
  WAIT_ASYNC_LE(P_STAGE_ASYNCS);
  __syncthreads();

  int buf = 0;
  for (int k0 = 0; k0 < CDIM; k0 += BK) {
    int bnext = buf + 2; if (bnext >= 3) bnext -= 3;
    stage(bnext, (k0 + 2 * BK) & (CDIM - 1));

    const __bf16* sA = sbase + (unsigned)buf * (P_BUF_BYTES / 2) + P_A_OFF / 2;
    const __bf16* sB = sbase + (unsigned)buf * (P_BUF_BYTES / 2) + P_B_OFF / 2;

    Frag16 fa[2], fb[2];
#pragma unroll
    for (int i = 0; i < 2; ++i) {
      int r = wm * 32 + i * 16 + lrow;
      fa[i].q[0] = *(const uint4*)&sA[r * PADK + lhi * 8];
      fa[i].q[1] = *(const uint4*)&sA[r * PADK + 16 + lhi * 8];
    }
#pragma unroll
    for (int j = 0; j < 2; ++j) {
      int r = wn * 32 + j * 16 + lrow;
      fb[j].q[0] = *(const uint4*)&sB[r * PADK + lhi * 8];
      fb[j].q[1] = *(const uint4*)&sB[r * PADK + 16 + lhi * 8];
    }
#pragma unroll
    for (int i = 0; i < 2; ++i)
#pragma unroll
      for (int j = 0; j < 2; ++j)
        acc[i][j] = __builtin_amdgcn_wmma_f32_16x16x32_bf16(
            false, fa[i].v, false, fb[j].v, (short)0, acc[i][j], false, false);

    WAIT_ASYNC_LE(P_STAGE_ASYNCS);
    __syncthreads();
    buf = buf + 1; if (buf >= 3) buf -= 3;
  }

#pragma unroll
  for (int i = 0; i < 2; ++i)
#pragma unroll
    for (int j = 0; j < 2; ++j)
#pragma unroll
      for (int e = 0; e < 8; ++e) {
        long row = m0 + wm * 32 + i * 16 + lhi * 8 + e;
        long col = n0 + wn * 32 + j * 16 + lrow;
        Out[row * CDIM + col] = acc[i][j][e];
      }
}

// ---------------------------------------------------------------------------
// Host orchestration
// ---------------------------------------------------------------------------
extern "C" void kernel_launch(void* const* d_in, const int* in_sizes, int n_in,
                              void* d_out, int out_size, void* d_ws, size_t ws_size,
                              hipStream_t stream) {
  const float* x  = (const float*)d_in[0];
  const float* wf = (const float*)d_in[1];
  const float* wg = (const float*)d_in[2];
  const float* wp = (const float*)d_in[3];
  float* out = (float*)d_out;

  char* ws = (char*)d_ws;
  const size_t nx = (size_t)MROWS * CDIM;           // 16,777,216
  const size_t nw = (size_t)CDIM * CDIM;            //  1,048,576

  size_t off = 0;
  __bf16* xb  = (__bf16*)(ws + off); off += nx * 2;         // 32 MB
  __bf16* wfb = (__bf16*)(ws + off); off += nw * 2;         //  2 MB
  __bf16* wgb = (__bf16*)(ws + off); off += nw * 2;
  __bf16* wpb = (__bf16*)(ws + off); off += nw * 2;
  __bf16* ab  = (__bf16*)(ws + off); off += nx * 2;         // 32 MB (a = f)
  __bf16* bb  = (__bf16*)(ws + off); off += nx * 2;         // 32 MB (b = (1-f)g)
  __bf16* hb  = (__bf16*)(ws + off); off += nx * 2;         // 32 MB
  float* aggP = (float*)(ws + off);  off += (size_t)NCHUNK * NCOLS * 4;
  float* aggQ = (float*)(ws + off);  off += (size_t)NCHUNK * NCOLS * 4;
  float* carry= (float*)(ws + off);  off += (size_t)NCHUNK * NCOLS * 4;
  (void)ws_size; (void)in_sizes; (void)n_in; (void)out_size;

  // 1) convert inputs to bf16
  cvt_f32_bf16<<<dim3((nx / 4 + 255) / 256), dim3(256), 0, stream>>>(x,  xb,  (int)nx);
  cvt_f32_bf16<<<dim3((nw / 4 + 255) / 256), dim3(256), 0, stream>>>(wf, wfb, (int)nw);
  cvt_f32_bf16<<<dim3((nw / 4 + 255) / 256), dim3(256), 0, stream>>>(wg, wgb, (int)nw);
  cvt_f32_bf16<<<dim3((nw / 4 + 255) / 256), dim3(256), 0, stream>>>(wp, wpb, (int)nw);

  // 2) fused gate GEMM -> a, b   (76800B dynamic LDS: 3 x [A|F|G])
  gemm_gates<<<dim3(MROWS / BM, CDIM / BN), dim3(256), 3 * G_BUF_BYTES, stream>>>(
      xb, wfb, wgb, ab, bb);

  // 3-5) chunked scan
  scan_pass1<<<dim3(NCOLS * NCHUNK / 256), dim3(256), 0, stream>>>(ab, bb, aggP, aggQ);
  scan_pass2<<<dim3(NCOLS / 256), dim3(256), 0, stream>>>(aggP, aggQ, carry);
  scan_pass3<<<dim3(NCOLS * NCHUNK / 256), dim3(256), 0, stream>>>(ab, bb, carry, hb);

  // 6) projection GEMM -> d_out  (46080B dynamic LDS: 3 x [A|B])
  gemm_proj<<<dim3(MROWS / BM, CDIM / BN), dim3(256), 3 * P_BUF_BYTES, stream>>>(
      hb, wpb, out);
}
